// LightWeightMultiHeadSelfAttention_52707838656834
// MI455X (gfx1250) — matrix operand order repaired and checked by reading, not compiled, not run on hardware
//
#include <hip/hip_runtime.h>

#define B_ 16
#define C_ 128
#define H_ 56
#define HC_ 4
#define R_ 4
#define DH_ 32
#define FLAT_ 3136
#define KVS_ 14
#define KV_ 196
#define KVP 224               // KV padded to 7*32 for WMMA K-chunks
#define SCALE_ 0.17677669529663687f   // 32^-0.5

typedef __attribute__((ext_vector_type(16))) _Float16 v16h;
typedef __attribute__((ext_vector_type(8)))  float    v8f;

__device__ __forceinline__ v8f wmma_f16(v16h a, v16h b, v8f c) {
  // (neg_a, A, neg_b, B, c_mod, C, reuse_a, reuse_b)
  return __builtin_amdgcn_wmma_f32_16x16x32_f16(false, a, false, b, (short)0, c,
                                                false, false);
}

// ---------------------------------------------------------------------------
// Depthwise 4x4 / stride 4 / pad 1 conv for both K and V branches.
// Output layout: [B*KV, C] row-major (GEMM-ready).
// ---------------------------------------------------------------------------
__global__ void dwconv_kv(const float* __restrict__ x,
                          const float* __restrict__ kw, const float* __restrict__ kb,
                          const float* __restrict__ vw, const float* __restrict__ vb,
                          float* __restrict__ kc, float* __restrict__ vc) {
  int idx = blockIdx.x * blockDim.x + threadIdx.x;   // == (b*KV+kv)*C + c
  if (idx >= B_ * KV_ * C_) return;
  int c  = idx % C_;
  int kv = (idx / C_) % KV_;
  int b  = idx / (C_ * KV_);
  int ky = kv / KVS_, kx = kv % KVS_;
  int iy0 = ky * R_ - 1, ix0 = kx * R_ - 1;
  const float* xp = x + ((size_t)b * C_ + c) * (H_ * H_);
  float sk = 0.f, sv = 0.f;
#pragma unroll
  for (int ry = 0; ry < R_; ++ry) {
    int iy = iy0 + ry;
    if (iy < 0 || iy >= H_) continue;
#pragma unroll
    for (int rx = 0; rx < R_; ++rx) {
      int ix = ix0 + rx;
      if (ix < 0 || ix >= H_) continue;
      float xv = xp[iy * H_ + ix];
      sk += xv * kw[c * 16 + ry * 4 + rx];
      sv += xv * vw[c * 16 + ry * 4 + rx];
    }
  }
  kc[idx] = sk + kb[c];
  vc[idx] = sv + vb[c];
}

// ---------------------------------------------------------------------------
// out[M,N] = A[M,K] @ W[N,K]^T + bias[N], fp32 in/out, f16 WMMA compute.
// A_TRANS=true: A is x in [B, K(=C), FLAT] layout, logical row m=(b,f).
// Block tile 64x64, 256 threads (8 waves), each wave -> two 16x16 D tiles.
// ---------------------------------------------------------------------------
template <bool A_TRANS>
__global__ void gemm_bias_wmma(const float* __restrict__ A,
                               const float* __restrict__ W,
                               const float* __restrict__ bias,
                               float* __restrict__ out,
                               int M, int N, int K) {
  __shared__ _Float16 As[64 * 32];   // [row][k]
  __shared__ _Float16 Bs[32 * 64];   // [k][n]

  const int tid  = threadIdx.x;
  const int lane = tid & 31;
  const int wave = tid >> 5;     // 0..7
  const int wm   = wave >> 2;    // 0..1  (M sub-tile of 32)
  const int wn   = wave & 3;     // 0..3  (N sub-tile of 16)
  const int ml   = lane & 15;
  const bool hi  = lane >= 16;
  const int mBase = blockIdx.x * 64;
  const int nBase = blockIdx.y * 64;

  v8f acc0 = {};
  v8f acc1 = {};

  for (int k0 = 0; k0 < K; k0 += 32) {
    // --- stage A tile (64x32) as f16 ---
#pragma unroll
    for (int it = 0; it < 8; ++it) {
      int idx = tid + it * 256;
      if (A_TRANS) {
        int col = idx >> 6, row = idx & 63;      // consecutive lanes -> rows (contig f)
        int m = mBase + row;
        int bb = m / FLAT_, f = m % FLAT_;
        As[row * 32 + col] =
            (_Float16)A[((size_t)bb * K + (k0 + col)) * FLAT_ + f];
      } else {
        int row = idx >> 5, col = idx & 31;      // consecutive lanes -> contig k
        As[row * 32 + col] = (_Float16)A[(size_t)(mBase + row) * K + k0 + col];
      }
    }
    // --- stage B tile: Bs[k][n] = W[nBase+n][k0+k] ---
#pragma unroll
    for (int it = 0; it < 8; ++it) {
      int idx = tid + it * 256;
      int n = idx >> 5, kk = idx & 31;
      Bs[kk * 64 + n] = (_Float16)W[(size_t)(nBase + n) * K + k0 + kk];
    }
    __syncthreads();

    // --- gather fragments per ISA VGPR layout and issue WMMAs ---
    v16h a0, a1, bf;
#pragma unroll
    for (int i = 0; i < 16; ++i) {
      int ka = i + (i < 8 ? 0 : 8) + (hi ? 8 : 0);     // A 16x32 f16 layout
      a0[i] = As[(wm * 32 + ml) * 32 + ka];
      a1[i] = As[(wm * 32 + 16 + ml) * 32 + ka];
      int kb2 = i + (i < 8 ? 0 : 8) + (hi ? 8 : 0);    // B 32x16 f16 layout
      bf[i] = Bs[kb2 * 64 + wn * 16 + ml];
    }
    acc0 = wmma_f16(a0, bf, acc0);
    acc1 = wmma_f16(a1, bf, acc1);
    __syncthreads();
  }

  // --- epilogue: bias + store (C/D layout: row = r + hi*8, col = ml) ---
  int n = nBase + wn * 16 + ml;
  float bv = bias[n];
#pragma unroll
  for (int r = 0; r < 8; ++r) {
    int m0 = mBase + wm * 32 + r + (hi ? 8 : 0);
    out[(size_t)m0 * N + n] = acc0[r] + bv;
    out[(size_t)(m0 + 16) * N + n] = acc1[r] + bv;
  }
}

// ---------------------------------------------------------------------------
// Fused attention: per (b, h, 16-row q tile):
//   S = Q Kt (WMMA) -> scale + bias + mask -> softmax -> O = P V (WMMA)
// Block = 64 threads (2 waves), each wave owns one q tile. KV padded to 224.
// ---------------------------------------------------------------------------
__global__ void attention_wmma(const float* __restrict__ qws,
                               const float* __restrict__ kp,
                               const float* __restrict__ vp,
                               const float* __restrict__ attn_bias,
                               float* __restrict__ ctx) {
  __shared__ _Float16 Ks[KVP * DH_];        // [kv][d]
  __shared__ _Float16 Vs[KVP * DH_];        // [kv][d]
  __shared__ float    Ss[2][16 * KVP];      // per-wave score/prob rows

  const int tid  = threadIdx.x;   // 0..63
  const int lane = tid & 31;
  const int wave = tid >> 5;      // 0..1
  const int ml   = lane & 15;
  const bool hi  = lane >= 16;

  int qg = blockIdx.x % 98;
  int bh = blockIdx.x / 98;
  int h  = bh % HC_;
  int b  = bh / HC_;
  int f0 = (qg * 2 + wave) * 16;

  // stage K,V for this (b,h) as f16; zero-fill the 196..223 pad rows
  for (int idx = tid; idx < KVP * DH_; idx += 64) {
    int kv = idx / DH_, d = idx % DH_;
    _Float16 kf = (_Float16)0.f, vf = (_Float16)0.f;
    if (kv < KV_) {
      size_t base = ((size_t)(b * KV_ + kv)) * C_ + h * DH_ + d;
      kf = (_Float16)kp[base];
      vf = (_Float16)vp[base];
    }
    Ks[idx] = kf;
    Vs[idx] = vf;
  }
  __syncthreads();

  // Q fragment (16 f x 32 d), one build, reused for all 14 K tiles
  v16h aQ;
  {
    size_t qbase = ((size_t)b * FLAT_ + f0 + ml) * C_ + h * DH_;
#pragma unroll
    for (int i = 0; i < 16; ++i) {
      int d = i + (i < 8 ? 0 : 8) + (hi ? 8 : 0);
      aQ[i] = (_Float16)qws[qbase + d];
    }
  }

  // phase 1: raw scores S[f][kv] = q . k
  float* S = Ss[wave];
  for (int t = 0; t < KVP / 16; ++t) {
    v16h bK;
#pragma unroll
    for (int i = 0; i < 16; ++i) {
      int d = i + (i < 8 ? 0 : 8) + (hi ? 8 : 0); // B k-dim = head dim
      bK[i] = Ks[(t * 16 + ml) * DH_ + d];       // n = kv = ml
    }
    v8f sc = {};
    sc = wmma_f16(aQ, bK, sc);
#pragma unroll
    for (int r = 0; r < 8; ++r) {
      int fl = r + (hi ? 8 : 0);
      S[fl * KVP + t * 16 + ml] = sc[r];
    }
  }
  __syncthreads();

  // phase 2: scale + attn_bias + pad mask, rowwise softmax.
  // lane handles row=ml, half=hi; halves combined with shfl_xor(16).
  const int row = ml;
  const int c0  = hi ? (KVP / 2) : 0;
  float mx = -3.0e38f;
  for (int j = 0; j < KVP / 2; ++j) {
    int kv = c0 + j;
    float s;
    if (kv < KV_) {
      s = S[row * KVP + kv] * SCALE_ +
          attn_bias[((size_t)h * FLAT_ + (f0 + row)) * KV_ + kv];
    } else {
      s = -1.0e30f;                              // mask pad columns
    }
    S[row * KVP + kv] = s;
    mx = fmaxf(mx, s);
  }
  mx = fmaxf(mx, __shfl_xor(mx, 16));
  float sum = 0.f;
  for (int j = 0; j < KVP / 2; ++j) {
    int kv = c0 + j;
    float e = __expf(S[row * KVP + kv] - mx);
    S[row * KVP + kv] = e;
    sum += e;
  }
  sum += __shfl_xor(sum, 16);
  float rowInv = 1.0f / sum;                     // lives in the a-frag row lane
  __syncthreads();

  // phase 3: O = P @ V over 7 K-chunks of 32, two N tiles (d 0..15, 16..31)
  v8f o0 = {};
  v8f o1 = {};
  for (int cc = 0; cc < KVP / 32; ++cc) {
    v16h aP, bV0, bV1;
#pragma unroll
    for (int i = 0; i < 16; ++i) {
      int k  = cc * 32 + i + (i < 8 ? 0 : 8) + (hi ? 8 : 0);
      aP[i]  = (_Float16)(S[ml * KVP + k] * rowInv);
      int kr = cc * 32 + i + (i < 8 ? 0 : 8) + (hi ? 8 : 0);
      bV0[i] = Vs[kr * DH_ + ml];
      bV1[i] = Vs[kr * DH_ + 16 + ml];
    }
    o0 = wmma_f16(aP, bV0, o0);
    o1 = wmma_f16(aP, bV1, o1);
  }

  // ctx in [B, FLAT, C] layout (ready for output projection)
#pragma unroll
  for (int r = 0; r < 8; ++r) {
    int f = f0 + r + (hi ? 8 : 0);
    size_t base = ((size_t)b * FLAT_ + f) * C_ + h * DH_;
    ctx[base + ml]      = o0[r];
    ctx[base + 16 + ml] = o1[r];
  }
}

// ---------------------------------------------------------------------------
extern "C" void kernel_launch(void* const* d_in, const int* in_sizes, int n_in,
                              void* d_out, int out_size, void* d_ws, size_t ws_size,
                              hipStream_t stream) {
  (void)in_sizes; (void)n_in; (void)out_size; (void)ws_size;
  const float* x   = (const float*)d_in[0];
  const float* kdw = (const float*)d_in[1];
  const float* kdb = (const float*)d_in[2];
  const float* vdw = (const float*)d_in[3];
  const float* vdb = (const float*)d_in[4];
  const float* wq  = (const float*)d_in[5];
  const float* bq  = (const float*)d_in[6];
  const float* wk  = (const float*)d_in[7];
  const float* bk  = (const float*)d_in[8];
  const float* wv  = (const float*)d_in[9];
  const float* bv  = (const float*)d_in[10];
  const float* wo  = (const float*)d_in[11];
  const float* bo  = (const float*)d_in[12];
  const float* ab  = (const float*)d_in[13];
  float* out = (float*)d_out;

  float* ws = (float*)d_ws;
  const size_t MQ = (size_t)B_ * FLAT_;   // 50176
  const size_t MK = (size_t)B_ * KV_;     // 3136
  float* q   = ws;  ws += MQ * C_;
  float* ctx = ws;  ws += MQ * C_;
  float* kc  = ws;  ws += MK * C_;
  float* vc  = ws;  ws += MK * C_;
  float* kpj = ws;  ws += MK * C_;
  float* vpj = ws;  ws += MK * C_;

  dwconv_kv<<<(B_ * KV_ * C_ + 255) / 256, 256, 0, stream>>>(
      x, kdw, kdb, vdw, vdb, kc, vc);

  gemm_bias_wmma<false><<<dim3((unsigned)(MK / 64), C_ / 64), 256, 0, stream>>>(
      kc, wk, bk, kpj, (int)MK, C_, C_);
  gemm_bias_wmma<false><<<dim3((unsigned)(MK / 64), C_ / 64), 256, 0, stream>>>(
      vc, wv, bv, vpj, (int)MK, C_, C_);
  gemm_bias_wmma<true><<<dim3((unsigned)(MQ / 64), C_ / 64), 256, 0, stream>>>(
      x, wq, bq, q, (int)MQ, C_, C_);

  attention_wmma<<<B_ * HC_ * 98, 64, 0, stream>>>(q, kpj, vpj, ab, ctx);

  gemm_bias_wmma<false><<<dim3((unsigned)(MQ / 64), C_ / 64), 256, 0, stream>>>(
      ctx, wo, bo, out, (int)MQ, C_, C_);
}
